// CausalSelfAttention2_66133906424390
// MI455X (gfx1250) — matrix operand order repaired and checked by reading, not compiled, not run on hardware
//
#include <hip/hip_runtime.h>
#include <hip/hip_bf16.h>

// Problem constants (match reference)
#define B_  4
#define T_  4096
#define C_  1024
#define H_  16
#define G_  8
#define GT_ 512
#define HS_ 64

typedef __attribute__((ext_vector_type(8)))  _Float16 v8h;
typedef __attribute__((ext_vector_type(16))) _Float16 v16h;
typedef __attribute__((ext_vector_type(8)))  float    v8f;

static __device__ __forceinline__ v16h cat8(v8h a, v8h b) {
  return __builtin_shufflevector(a, b, 0,1,2,3,4,5,6,7,8,9,10,11,12,13,14,15);
}
static __device__ __forceinline__ v8f wmma16(v16h a, v16h b, v8f c) {
  // D = A(16x32 f16) * B(32x16 f16) + C(16x16 f32)
  return __builtin_amdgcn_wmma_f32_16x16x32_f16(false, a, false, b, (short)0, c, false, false);
}

// ---------------------------------------------------------------- converts
__global__ void cvt_f32_to_f16(const float* __restrict__ s, _Float16* __restrict__ d, int n) {
  int i = blockIdx.x * blockDim.x + threadIdx.x;
  if (i < n) d[i] = (_Float16)s[i];
}

// src [rows][cols] f32 -> dst [cols][rows] f16  (so GEMM B-frags are contiguous)
__global__ void transpose_to_f16(const float* __restrict__ s, _Float16* __restrict__ d,
                                 int rows, int cols) {
  int i = blockIdx.x * blockDim.x + threadIdx.x;
  if (i < rows * cols) {
    int r = i / cols, c = i - r * cols;
    d[(size_t)c * rows + r] = (_Float16)s[i];
  }
}

// ---------------------------------------------------------------- WMMA GEMM
// C[M,N] = A[M,K](f16,row-major) * BT[N,K](f16,row-major == B^T)
// Block: 128 threads (4 waves). Wave tile: 32 rows x 64 cols.
template<bool F16OUT>
__global__ void __launch_bounds__(128)
gemm_wmma(const _Float16* __restrict__ A, const _Float16* __restrict__ BT,
          void* __restrict__ Cout, int M, int N, int K) {
  const int lane = threadIdx.x & 31;
  const int wave = threadIdx.x >> 5;
  const int l16  = lane & 15;
  const int hi   = lane >> 4;            // 0: lanes 0-15, 1: lanes 16-31
  const int mbase = blockIdx.x * 128 + wave * 32;
  const int nbase = blockIdx.y * 64;

  v8f acc[2][4];
#pragma unroll
  for (int a = 0; a < 2; ++a)
#pragma unroll
    for (int bn = 0; bn < 4; ++bn) acc[a][bn] = (v8f){0,0,0,0,0,0,0,0};

  const _Float16* arow0 = A + (size_t)(mbase +      l16) * K;  // A-frag: M = lane%16
  const _Float16* arow1 = A + (size_t)(mbase + 16 + l16) * K;
  const int akoff = hi * 16;     // contiguous k order used here (A): lanes>=16 hold K 16..31
  const int bkoff = hi * 16;     // same contiguous k order for B: lanes>=16 hold K 16..31

  for (int k0 = 0; k0 < K; k0 += 32) {
    v16h a0 = cat8(*(const v8h*)(arow0 + k0 + akoff), *(const v8h*)(arow0 + k0 + akoff + 8));
    v16h a1 = cat8(*(const v8h*)(arow1 + k0 + akoff), *(const v8h*)(arow1 + k0 + akoff + 8));
#pragma unroll
    for (int nt = 0; nt < 4; ++nt) {
      const _Float16* brow = BT + (size_t)(nbase + nt * 16 + l16) * K + k0 + bkoff;
      v16h b = *(const v16h*)brow;                 // 32B contiguous (B^T row)
      acc[0][nt] = wmma16(a0, b, acc[0][nt]);
      acc[1][nt] = wmma16(a1, b, acc[1][nt]);
    }
    if (k0 + 32 < K) __builtin_prefetch(arow0 + k0 + 32, 0, 0);
  }

#pragma unroll
  for (int rt = 0; rt < 2; ++rt)
#pragma unroll
    for (int nt = 0; nt < 4; ++nt)
#pragma unroll
      for (int r = 0; r < 8; ++r) {
        int row = mbase + rt * 16 + r + hi * 8;    // C layout: lanes>=16 -> M+8
        int col = nbase + nt * 16 + l16;
        if constexpr (F16OUT)
          ((_Float16*)Cout)[(size_t)row * N + col] = (_Float16)acc[rt][nt][r];
        else
          ((float*)Cout)[(size_t)row * N + col] = acc[rt][nt][r];
      }
}

// ---------------------------------------------------------------- RoPE + block scatter
// qkvh: [B*T, 3C] f16. Writes:
//   qh,kh: [B][H][G][GT][HS] f16 (raw-view-scrambled grouping of the reference)
//   vT:    [B][H][G][HS][GT] f16 (transposed so PV B-frags are contiguous)
__global__ void rope_scatter(const _Float16* __restrict__ qkvh,
                             const float* __restrict__ cosT, const float* __restrict__ sinT,
                             _Float16* __restrict__ qh, _Float16* __restrict__ kh,
                             _Float16* __restrict__ vT) {
  int idx = blockIdx.x * blockDim.x + threadIdx.x;   // B*H*T*32
  int i  = idx & 31;           // rotary pair index
  int t  = (idx >> 5) & (T_ - 1);
  int hp = (idx >> 17) & (H_ - 1);
  int b  = idx >> 21;

  size_t row = (size_t)b * T_ + t;
  const _Float16* qp = qkvh + row * (3 * C_) + hp * HS_ + 2 * i;
  const _Float16* kp = qp + C_;
  const _Float16* vp = qp + 2 * C_;

  float c = cosT[t * 32 + i], s = sinT[t * 32 + i];
  float qr = (float)qp[0], qi = (float)qp[1];
  float kr = (float)kp[0], ki = (float)kp[1];
  float qro = qr * c - qi * s, qio = qr * s + qi * c;
  float kro = kr * c - ki * s, kio = kr * s + ki * c;

  // raw .view reinterpretation: (b,h',t') -> block (b, hh, g) row gt
  int hh = t & 15;
  int g  = hp >> 1;
  int gt = ((hp & 1) << 8) | (t >> 4);
  size_t qo = ((((size_t)b * H_ + hh) * G_ + g) * GT_ + gt) * HS_ + 2 * i;
  qh[qo] = (_Float16)qro; qh[qo + 1] = (_Float16)qio;
  kh[qo] = (_Float16)kro; kh[qo + 1] = (_Float16)kio;

  // v: natural grouping (v was never head-transposed); store transposed [d][gt]
  int gv = t >> 9, gtv = t & 511;
  size_t vb = (((size_t)b * H_ + hp) * G_ + gv) * (size_t)(HS_ * GT_);
  vT[vb + (size_t)(2 * i)     * GT_ + gtv] = vp[0];
  vT[vb + (size_t)(2 * i + 1) * GT_ + gtv] = vp[1];
}

// ---------------------------------------------------------------- time pooling (qb,kb,vb)
__global__ void pool_kernel(const _Float16* __restrict__ qh, const _Float16* __restrict__ kh,
                            const _Float16* __restrict__ vT,
                            const float* __restrict__ qproj, const float* __restrict__ kproj,
                            const float* __restrict__ vproj, float* __restrict__ outb) {
  int idx = blockIdx.x * blockDim.x + threadIdx.x;   // B*H*(G-1)*HS = 28672
  if (idx >= B_ * H_ * (G_ - 1) * HS_) return;
  int d = idx & 63;
  int g = (idx >> 6) % (G_ - 1);
  int h = (idx / (64 * (G_ - 1))) % H_;
  int b = idx / (64 * (G_ - 1) * H_);

  const _Float16* q = qh + ((((size_t)b * H_ + h) * G_ + g) * GT_) * HS_ + d;
  const _Float16* k = kh + ((((size_t)b * H_ + h) * G_ + g) * GT_) * HS_ + d;
  const _Float16* v = vT + ((((size_t)b * H_ + h) * G_ + g) * HS_ + d) * (size_t)GT_;
  float aq = 0.f, ak = 0.f, av = 0.f;
  for (int t = 0; t < GT_; ++t) {
    aq += (float)q[(size_t)t * HS_] * qproj[t];
    ak += (float)k[(size_t)t * HS_] * kproj[t];
    av += (float)v[t] * vproj[t];
  }
  size_t o = (((size_t)b * H_ + h) * (G_ - 1) + g) * HS_ + d;
  const size_t OUT0 = (size_t)B_ * T_ * C_;          // 16777216
  const size_t PB   = (size_t)B_ * H_ * (G_ - 1) * HS_;
  outb[OUT0 + o]          = aq;
  outb[OUT0 + PB + o]     = ak;
  outb[OUT0 + 2 * PB + o] = av;   // causal_attn on length-1 blocks is identity
}

// ---------------------------------------------------------------- flash attention per block
// grid: B*H*G*8 blocks; block: 128 threads (4 waves x 16 query rows)
__global__ void __launch_bounds__(128)
attn_kernel(const _Float16* __restrict__ qh, const _Float16* __restrict__ kh,
            const _Float16* __restrict__ vT, _Float16* __restrict__ oh) {
  __shared__ __align__(16) _Float16 plds[4][16][72];   // per-wave P staging (C->A relayout)

  const int lane = threadIdx.x & 31;
  const int wave = threadIdx.x >> 5;
  const int l16  = lane & 15;
  const int hi   = lane >> 4;
  const int qt = blockIdx.x & 7;
  const int g  = (blockIdx.x >> 3) & 7;
  const int h  = (blockIdx.x >> 6) & 15;
  const int b  = blockIdx.x >> 10;

  const size_t blk = ((size_t)b * H_ + h) * G_ + g;
  const _Float16* Q  = qh + blk * (size_t)(GT_ * HS_);
  const _Float16* Kb = kh + blk * (size_t)(GT_ * HS_);
  const _Float16* Vb = vT + blk * (size_t)(HS_ * GT_);   // [d][t]

  const int akoff = hi * 16;
  const int bkoff = hi * 16;

  // Q A-fragments for this wave's 16 rows (HS=64 -> two K=32 frags)
  const _Float16* qp = Q + (size_t)(qt * 64 + wave * 16 + l16) * HS_;
  v16h aq0 = cat8(*(const v8h*)(qp + akoff),      *(const v8h*)(qp + akoff + 8));
  v16h aq1 = cat8(*(const v8h*)(qp + 32 + akoff), *(const v8h*)(qp + 32 + akoff + 8));

  float mi[8], li[8];
  v8f acc[4];
#pragma unroll
  for (int r = 0; r < 8; ++r) { mi[r] = -1e30f; li[r] = 0.f; }
#pragma unroll
  for (int dt = 0; dt < 4; ++dt) acc[dt] = (v8f){0,0,0,0,0,0,0,0};

  for (int c = 0; c <= qt; ++c) {
    const int kb0 = c * 64;
    if (c < qt) __builtin_prefetch(Kb + (size_t)(kb0 + 64 + lane) * HS_, 0, 0);

    // ---- S = Q K^T for 4 key sub-tiles of 16
    v8f sfr[4];
#pragma unroll
    for (int st = 0; st < 4; ++st) {
      const _Float16* kp = Kb + (size_t)(kb0 + st * 16 + l16) * HS_;
      v16h b0 = *(const v16h*)(kp + bkoff);        // d = 0..31 chunk
      v16h b1 = *(const v16h*)(kp + 32 + bkoff);   // d = 32..63 chunk
      v8f sv = (v8f){0,0,0,0,0,0,0,0};
      sv = wmma16(aq0, b0, sv);
      sv = wmma16(aq1, b1, sv);
      sfr[st] = sv;
    }

    // ---- scale, causal mask (diagonal chunk), row max
    const bool diag = (c == qt);
    float mnew[8];
#pragma unroll
    for (int r = 0; r < 8; ++r) {
      const int m = qt * 64 + wave * 16 + r + hi * 8;   // query row (local)
      float mx = mi[r];
#pragma unroll
      for (int st = 0; st < 4; ++st) {
        float v = sfr[st][r] * 0.125f;                   // 1/sqrt(64)
        int n = kb0 + st * 16 + l16;                     // key (local)
        if (diag && n > m) v = -1e30f;
        sfr[st][r] = v;
        mx = fmaxf(mx, v);
      }
      mx = fmaxf(mx, __shfl_xor(mx, 1, 32));
      mx = fmaxf(mx, __shfl_xor(mx, 2, 32));
      mx = fmaxf(mx, __shfl_xor(mx, 4, 32));
      mx = fmaxf(mx, __shfl_xor(mx, 8, 32));
      mnew[r] = mx;
    }

    // ---- exp, write P (f16) to LDS, online rescale
#pragma unroll
    for (int r = 0; r < 8; ++r) {
      float cf = __expf(mi[r] - mnew[r]);
      float ps = 0.f;
#pragma unroll
      for (int st = 0; st < 4; ++st) {
        float p = __expf(sfr[st][r] - mnew[r]);
        ps += p;
        plds[wave][r + hi * 8][st * 16 + l16] = (_Float16)p;
      }
      ps += __shfl_xor(ps, 1, 32);
      ps += __shfl_xor(ps, 2, 32);
      ps += __shfl_xor(ps, 4, 32);
      ps += __shfl_xor(ps, 8, 32);
      li[r] = li[r] * cf + ps;
      mi[r] = mnew[r];
#pragma unroll
      for (int dt = 0; dt < 4; ++dt) acc[dt][r] *= cf;
    }
    asm volatile("s_wait_dscnt 0" ::: "memory");   // cross-lane LDS visibility in-wave

    // ---- read P back as A-fragments (rows = lane%16)
    const _Float16* pr = &plds[wave][l16][0];
    v16h ap0 = cat8(*(const v8h*)(pr + akoff),      *(const v8h*)(pr + akoff + 8));
    v16h ap1 = cat8(*(const v8h*)(pr + 32 + akoff), *(const v8h*)(pr + 32 + akoff + 8));

    // ---- O += P V  (K-dim = 64 keys -> 2 WMMAs per 16-wide d tile)
#pragma unroll
    for (int dt = 0; dt < 4; ++dt) {
      const _Float16* vp = Vb + (size_t)(dt * 16 + l16) * GT_ + kb0;  // V^T row = output dim
      v16h bv0 = *(const v16h*)(vp + bkoff);
      v16h bv1 = *(const v16h*)(vp + 32 + bkoff);
      acc[dt] = wmma16(ap0, bv0, acc[dt]);
      acc[dt] = wmma16(ap1, bv1, acc[dt]);
    }
  }

  // ---- epilogue: normalize, store f16 into [B*T, C] for the output GEMM
#pragma unroll
  for (int r = 0; r < 8; ++r) {
    float inv = 1.0f / li[r];
    int gt = qt * 64 + wave * 16 + r + hi * 8;
    size_t row = (size_t)b * T_ + (size_t)g * GT_ + gt;
#pragma unroll
    for (int dt = 0; dt < 4; ++dt)
      oh[row * C_ + h * HS_ + dt * 16 + l16] = (_Float16)(acc[dt][r] * inv);
  }
}

// ---------------------------------------------------------------- launcher
extern "C" void kernel_launch(void* const* d_in, const int* in_sizes, int n_in,
                              void* d_out, int out_size, void* d_ws, size_t ws_size,
                              hipStream_t stream) {
  const float* x     = (const float*)d_in[0];
  const float* Wqkv  = (const float*)d_in[1];
  const float* Wo    = (const float*)d_in[2];
  const float* qproj = (const float*)d_in[3];
  const float* kproj = (const float*)d_in[4];
  const float* vproj = (const float*)d_in[5];
  const float* fcos  = (const float*)d_in[6];
  const float* fsin  = (const float*)d_in[7];
  float* out = (float*)d_out;

  unsigned char* ws = (unsigned char*)d_ws;
  const size_t SZ_XH    = (size_t)B_ * T_ * C_ * 2;        // 32MB  f16 x
  const size_t SZ_WQKVT = (size_t)3 * C_ * C_ * 2;         // 6MB   f16 Wqkv^T
  const size_t SZ_WOT   = (size_t)C_ * C_ * 2;             // 2MB   f16 Wo^T
  const size_t SZ_QKV   = (size_t)B_ * T_ * 3 * C_ * 2;    // 96MB  f16 qkv
  const size_t SZ_BLK   = (size_t)B_ * H_ * G_ * GT_ * HS_ * 2; // 32MB each

  _Float16* xh     = (_Float16*)(ws);
  _Float16* wqkvT  = (_Float16*)(ws + SZ_XH);
  _Float16* woT    = (_Float16*)(ws + SZ_XH + SZ_WQKVT);
  _Float16* qkvh   = (_Float16*)(ws + SZ_XH + SZ_WQKVT + SZ_WOT);
  _Float16* qh     = (_Float16*)(ws + SZ_XH + SZ_WQKVT + SZ_WOT + SZ_QKV);
  _Float16* kh     = (_Float16*)((unsigned char*)qh + SZ_BLK);
  _Float16* vT     = (_Float16*)((unsigned char*)kh + SZ_BLK);
  _Float16* oh     = qkvh;   // alias: qkv consumed by rope_scatter before attention writes

  // 1) converts
  {
    int n = B_ * T_ * C_;
    cvt_f32_to_f16<<<(n + 255) / 256, 256, 0, stream>>>(x, xh, n);
    transpose_to_f16<<<(C_ * 3 * C_ + 255) / 256, 256, 0, stream>>>(Wqkv, wqkvT, C_, 3 * C_);
    transpose_to_f16<<<(C_ * C_ + 255) / 256, 256, 0, stream>>>(Wo, woT, C_, C_);
  }
  // 2) QKV GEMM: [16384,1024] x [1024,3072] -> f16
  {
    dim3 grid((B_ * T_) / 128, (3 * C_) / 64);
    gemm_wmma<true><<<grid, 128, 0, stream>>>(xh, wqkvT, qkvh, B_ * T_, 3 * C_, C_);
  }
  // 3) RoPE + scatter into blocked layouts
  {
    int n = B_ * H_ * T_ * 32;
    rope_scatter<<<n / 256, 256, 0, stream>>>(qkvh, fcos, fsin, qh, kh, vT);
  }
  // 4) pooled qb/kb/vb outputs
  {
    int n = B_ * H_ * (G_ - 1) * HS_;
    pool_kernel<<<(n + 255) / 256, 256, 0, stream>>>(qh, kh, vT, qproj, kproj, vproj, out);
  }
  // 5) block-local causal flash attention
  {
    attn_kernel<<<B_ * H_ * G_ * 8, 128, 0, stream>>>(qh, kh, vT, oh);
  }
  // 6) output GEMM: [16384,1024] x [1024,1024] -> f32 d_out
  {
    dim3 grid((B_ * T_) / 128, C_ / 64);
    gemm_wmma<false><<<grid, 128, 0, stream>>>(oh, woT, out, B_ * T_, C_, C_);
  }
}
